// HyperMLPHead_38637525795460
// MI455X (gfx1250) — matrix operand order repaired and checked by reading, not compiled, not run on hardware
//
#include <hip/hip_runtime.h>
#include <math.h>

// ---------------------------------------------------------------------------
// Types
// ---------------------------------------------------------------------------
typedef __bf16 bf16;
typedef __attribute__((ext_vector_type(16))) __bf16 v16bf;
typedef __attribute__((ext_vector_type(8)))  __bf16 v8bf;
typedef __attribute__((ext_vector_type(8)))  float  v8f;
typedef __attribute__((ext_vector_type(4)))  int    v4i;

// pointer-to-int4 in global (1) / LDS (3) address spaces, as expected by
// __builtin_amdgcn_global_load_async_to_lds_b128 (per hipcc diagnostic).
typedef __attribute__((address_space(1))) v4i* as1_v4i;
typedef __attribute__((address_space(3))) v4i* as3_v4i;

#if defined(__AMDGCN__) && __has_builtin(__builtin_amdgcn_global_load_async_to_lds_b128)
#define GEMM_ASYNC 1
#else
#define GEMM_ASYNC 0
#endif

__device__ __forceinline__ void wait_async0()
{
#if defined(__AMDGCN__)
#if __has_builtin(__builtin_amdgcn_s_wait_asynccnt)
    __builtin_amdgcn_s_wait_asynccnt(0);
#else
    asm volatile("s_wait_asynccnt 0x0" ::: "memory");
#endif
#endif
}

// ---------------------------------------------------------------------------
// Problem constants (from reference): B,T,D,H,DH,RF,RS
// ---------------------------------------------------------------------------
constexpr int CB  = 2;
constexpr int CT  = 1024;
constexpr int CD  = 1024;
constexpr int CH  = 16;
constexpr int CDH = 64;
constexpr int CRF = 128;
constexpr int CRS = 32;
constexpr int CHD = CH * CDH;                 // 1024

constexpr long NX = (long)CB * CT * CD;       // x elements           (2M)
constexpr long NF = (long)CB * CH * CT * CDH; // one feature tensor   (2M)
constexpr long NZ = (long)CB * CH * CT * CT;  // one lag tensor       (32M)

// ---------------------------------------------------------------------------
// Generic batched GEMM:  C[b] = A[b] (MxK) * B[b] (KxN), bf16 inputs.
//   A: element (m,k) at m*a_rs + k          (K contiguous)
//   B: element (k,n) at n*b_ns + k          (K contiguous; callers pass B^T)
//   C: c_mode 0: f32, C[m*c_rs+n]; 1: bf16, C[m*c_rs+n]; 2: bf16, C[n*c_rs+m]
// Block: 128 threads = 4 waves; block tile 64x64; each wave 32x32
// (2x2 of 16x16 WMMA tiles); K staged 64 per pass = 2 WMMA K-steps.
// Double-buffered LDS + software pipeline: next tile is fetched (async
// global->LDS when available, else into registers) while WMMAs run on the
// current tile; two barriers per K iteration.
// OOB columns (N<64) are clamped, their garbage only reaches discarded C
// columns; OOB K (K=32) skips the second WMMA half-step (uniform branch).
// ---------------------------------------------------------------------------
__global__ __launch_bounds__(128)
void gemm_bf16_kernel(const bf16* __restrict__ A, const bf16* __restrict__ Bm,
                      void* __restrict__ Cv, int c_mode,
                      int M, int N, int K,
                      long a_rs, long b_ns, long c_rs,
                      long a_bs, long b_bs, long c_bs)
{
    __shared__ bf16 As[2][64][80] __attribute__((aligned(64)));
    __shared__ bf16 Bs[2][64][80] __attribute__((aligned(64)));   // Bs[n][k]

    const int tid  = threadIdx.x;
    const int lane = tid & 31;
    const int wid  = tid >> 5;
    const int half = lane >> 4;
    const int l16  = lane & 15;
    const int m0   = blockIdx.y * 64;
    const int n0   = blockIdx.x * 64;
    const long bat = blockIdx.z;

    const bf16* Ab = A  + bat * a_bs;
    const bf16* Bb = Bm + bat * b_bs;

    const int wrow = (wid >> 1) * 32;
    const int wcol = (wid & 1) * 32;

    // staging coordinates: 512 v8bf vectors per matrix, 4 per thread
    int sr[4], sc[4];
    const bf16* pa[4];
    const bf16* pb[4];
#pragma unroll
    for (int i = 0; i < 4; ++i) {
        int lin = tid + 128 * i;          // 0..511
        sr[i] = lin >> 3;                 // A row / B column, 0..63
        sc[i] = (lin & 7) << 3;           // k offset, 0..56
        int gn  = n0 + sr[i];
        int gnc = gn < N ? gn : N - 1;    // clamp; garbage cols are discarded
        pa[i] = Ab + (long)(m0 + sr[i]) * a_rs;
        pb[i] = Bb + (long)gnc * b_ns;
    }
    auto kcl = [&](int i, int k0) -> long {
        long ko = k0 + sc[i];
        return ko < K ? ko : 0;           // clamp; tail never consumed
    };

#if GEMM_ASYNC
    auto issue_tile = [&](int buf, int k0) {
#pragma unroll
        for (int i = 0; i < 4; ++i) {
            long kc = kcl(i, k0);
            __builtin_amdgcn_global_load_async_to_lds_b128(
                (as1_v4i)(unsigned long long)(pa[i] + kc),
                (as3_v4i)(unsigned long long)&As[buf][sr[i]][sc[i]], 0, 0);
            __builtin_amdgcn_global_load_async_to_lds_b128(
                (as1_v4i)(unsigned long long)(pb[i] + kc),
                (as3_v4i)(unsigned long long)&Bs[buf][sr[i]][sc[i]], 0, 0);
        }
    };
    issue_tile(0, 0);
#else
    v8bf va[4], vb[4];
    auto load_tile = [&](int k0) {
#pragma unroll
        for (int i = 0; i < 4; ++i) {
            long kc = kcl(i, k0);
            va[i] = *(const v8bf*)(pa[i] + kc);
            vb[i] = *(const v8bf*)(pb[i] + kc);
        }
    };
    auto store_tile = [&](int buf) {
#pragma unroll
        for (int i = 0; i < 4; ++i) {
            *(v8bf*)&As[buf][sr[i]][sc[i]] = va[i];
            *(v8bf*)&Bs[buf][sr[i]][sc[i]] = vb[i];
        }
    };
    load_tile(0);
#endif

    v8f acc[2][2];
    const v8f vzero = {0.f, 0.f, 0.f, 0.f, 0.f, 0.f, 0.f, 0.f};
#pragma unroll
    for (int mi = 0; mi < 2; ++mi)
#pragma unroll
        for (int ni = 0; ni < 2; ++ni)
            acc[mi][ni] = vzero;

    for (int k0 = 0; k0 < K; k0 += 64) {
        const int buf = (k0 >> 6) & 1;
#if GEMM_ASYNC
        wait_async0();                    // own wave's copies to LDS done
        __syncthreads();                  // all waves' data visible; prev compute done
        if (k0 + 64 < K) issue_tile(buf ^ 1, k0 + 64);
#else
        store_tile(buf);
        __syncthreads();                  // tile visible; prev compute done
        if (k0 + 64 < K) {
            load_tile(k0 + 64);           // overlap global latency with WMMAs
            if (k0 + 128 < K) {           // gfx1250 global_prefetch_b8
                __builtin_prefetch(pa[0] + (k0 + 128), 0, 1);
                __builtin_prefetch(pb[0] + (k0 + 128), 0, 1);
            }
        }
#endif

#pragma unroll
        for (int kk = 0; kk < 64; kk += 32) {
            if (k0 + kk < K) {            // uniform; skips tail when K==32
#pragma unroll
                for (int mi = 0; mi < 2; ++mi) {
                    // A fragment (16x32): lanes 0-15 rows M, K {0..7,16..23};
                    // lanes 16-31 same rows, K {8..15,24..31}.
                    int arow = wrow + mi * 16 + l16;
                    v8bf alo = *(const v8bf*)&As[buf][arow][kk + half * 8];
                    v8bf ahi = *(const v8bf*)&As[buf][arow][kk + 16 + half * 8];
                    v16bf af;
#pragma unroll
                    for (int e = 0; e < 8; ++e) { af[e] = alo[e]; af[e + 8] = ahi[e]; }

#pragma unroll
                    for (int ni = 0; ni < 2; ++ni) {
                        // B fragment (32x16): lane = column; lanes 0-15 K {0..7,16..23},
                        // lanes 16-31 K {8..15,24..31} (matches A).
                        int bcol = wcol + ni * 16 + l16;
                        v8bf blo = *(const v8bf*)&Bs[buf][bcol][kk + half * 8];
                        v8bf bhi = *(const v8bf*)&Bs[buf][bcol][kk + 16 + half * 8];
                        v16bf bfrag;
#pragma unroll
                        for (int e = 0; e < 8; ++e) { bfrag[e] = blo[e]; bfrag[e + 8] = bhi[e]; }
                        acc[mi][ni] = __builtin_amdgcn_wmma_f32_16x16x32_bf16(
                            false, af, false, bfrag, (short)0, acc[mi][ni], false, false);
                    }
                }
            }
        }
        __syncthreads();
    }

    // ---- store C: lane 0-15 -> N=lane, VGPR i -> M=i; lanes 16-31 -> M=8+i ----
    float* Cf = (float*)Cv + bat * c_bs;
    bf16*  Ch = (bf16*)Cv + bat * c_bs;
#pragma unroll
    for (int mi = 0; mi < 2; ++mi)
#pragma unroll
        for (int ni = 0; ni < 2; ++ni) {
            int col = n0 + wcol + ni * 16 + l16;
            if (col >= N) continue;
#pragma unroll
            for (int i = 0; i < 8; ++i) {
                int row = m0 + wrow + mi * 16 + half * 8 + i;
                float val = acc[mi][ni][i];
                if (c_mode == 0)      Cf[(long)row * c_rs + col] = val;
                else if (c_mode == 1) Ch[(long)row * c_rs + col] = (bf16)val;
                else                  Ch[(long)col * c_rs + row] = (bf16)val;
            }
        }
}

// ---------------------------------------------------------------------------
// Elementwise kernels
// ---------------------------------------------------------------------------
__global__ void cvt_bf16_kernel(const float* __restrict__ src, bf16* __restrict__ dst, long n)
{
    long i = (long)blockIdx.x * blockDim.x + threadIdx.x;
    if (i < n) dst[i] = (bf16)src[i];
}

// transpose-convert: src [rows][cols] f32 -> dst [cols][rows] bf16
__global__ void cvtT_bf16_kernel(const float* __restrict__ src, bf16* __restrict__ dst,
                                 int rows, int cols)
{
    long i = (long)blockIdx.x * blockDim.x + threadIdx.x;
    if (i >= (long)rows * cols) return;
    int c = (int)(i % cols);
    long r = i / cols;
    dst[(long)c * rows + r] = (bf16)src[i];
}

// Y: [8][B*T][H*DH] f32 -> gated bf16 feature tensors.
// QM/Kp in [B,H,T,DH]; Vfb transposed to [B,H,DH,T] (K-contiguous for ctx GEMM).
__global__ void prep_feats_kernel(const float* __restrict__ Y,
                                  bf16* __restrict__ QM, bf16* __restrict__ Kp,
                                  bf16* __restrict__ Vfb, float* __restrict__ MVO)
{
    long idx = (long)blockIdx.x * blockDim.x + threadIdx.x;
    if (idx >= NF) return;
    int d = (int)(idx % CDH); long r = idx / CDH;
    int t = (int)(r % CT);    r /= CT;
    int h = (int)(r % CH);
    int b = (int)(r / CH);
    long yoff = ((long)b * CT + t) * CHD + (long)h * CDH + d;
    const long ys = NF;
    float q   = Y[0 * ys + yoff], mg  = Y[1 * ys + yoff], k  = Y[2 * ys + yoff];
    float qs  = Y[3 * ys + yoff], mgs = Y[4 * ys + yoff], ks = Y[5 * ys + yoff];
    float v   = Y[6 * ys + yoff], mvo = Y[7 * ys + yoff];
    QM[idx]      = (bf16)(q  * mg);
    QM[NF + idx] = (bf16)(qs * mgs);
    Kp[idx]      = (bf16)k;
    Kp[NF + idx] = (bf16)ks;
    Vfb[(((long)b * CH + h) * CDH + d) * CT + t] = (bf16)v;   // [b,h,d,t]
    MVO[idx]     = mvo;
}

// S: [2,B,H,T,T] scores -> causal-lag bf16: z[t,s] = (s<=t) ? S[t,t-s] : 0
__global__ void lag_kernel(const float* __restrict__ S, bf16* __restrict__ Zb)
{
    long idx = (long)blockIdx.x * blockDim.x + threadIdx.x;
    if (idx >= 2 * NZ) return;
    int s = (int)(idx % CT); long r = idx / CT;
    int t = (int)(r % CT);   long bh = r / CT;
    float v = 0.f;
    if (s <= t) v = S[(bh * CT + t) * (long)CT + (t - s)];
    Zb[idx] = (bf16)v;
}

// out[t,s] = tril( z[t,s]*lam[s] + Q2[t,s] ).
// gather==1: z re-gathered from the score tensor (lag layout on the fly);
// gather==0: src already in lag layout.
__global__ void seqmix_combine_kernel(const float* __restrict__ src, int gather,
                                      const float* __restrict__ Q2,
                                      const float* __restrict__ lam,
                                      float* __restrict__ outf, bf16* __restrict__ outb)
{
    long idx = (long)blockIdx.x * blockDim.x + threadIdx.x;
    if (idx >= NZ) return;
    int s = (int)(idx % CT); long r = idx / CT;
    int t = (int)(r % CT);   long bh = r / CT;
    float v = 0.f;
    if (s <= t) {
        float zv = gather ? src[(bh * CT + t) * (long)CT + (t - s)] : src[idx];
        v = zv * lam[s] + Q2[idx];
    }
    if (outf) outf[idx] = v;
    if (outb) outb[idx] = (bf16)v;
}

// L2-normalize each length-T row (h1 / (||h1|| + 1e-8)); one block per row.
__global__ __launch_bounds__(256) void rownorm_kernel(float* __restrict__ Hm)
{
    long row = blockIdx.x;
    float* p = Hm + row * (long)CT;
    __shared__ float red[256];
    float ss = 0.f;
    for (int i = threadIdx.x; i < CT; i += 256) { float v = p[i]; ss += v * v; }
    red[threadIdx.x] = ss;
    __syncthreads();
    for (int o = 128; o > 0; o >>= 1) {
        if ((int)threadIdx.x < o) red[threadIdx.x] += red[threadIdx.x + o];
        __syncthreads();
    }
    float inv = 1.f / (sqrtf(red[0]) + 1e-8f);
    for (int i = threadIdx.x; i < CT; i += 256) p[i] *= inv;
}

// h = softplus(h1s) * relu(h1n)
__global__ void gate_kernel(const float* __restrict__ H1n, const float* __restrict__ H1s,
                            float* __restrict__ Hf, bf16* __restrict__ Hb)
{
    long idx = (long)blockIdx.x * blockDim.x + threadIdx.x;
    if (idx >= NZ) return;
    float a = H1n[idx], c = H1s[idx];
    float sp = (c > 20.f) ? c : log1pf(expf(c));
    float re = (a > 0.f) ? a : 0.f;
    float v = sp * re;
    Hf[idx] = v;
    Hb[idx] = (bf16)v;
}

// inter = ctx * mvo, packed [B*T, H*DH] bf16 for the output projection
__global__ void inter_kernel(const float* __restrict__ ctxb, const float* __restrict__ MVO,
                             bf16* __restrict__ interb)
{
    long idx = (long)blockIdx.x * blockDim.x + threadIdx.x;
    if (idx >= NF) return;
    int d = (int)(idx % CDH); long r = idx / CDH;
    int t = (int)(r % CT);    r /= CT;
    int h = (int)(r % CH);
    int b = (int)(r / CH);
    float v = ctxb[idx] * MVO[idx];
    interb[((long)b * CT + t) * CHD + (long)h * CDH + d] = (bf16)v;
}

// ---------------------------------------------------------------------------
// Host orchestration
// ---------------------------------------------------------------------------
static void launch_gemm(hipStream_t s, const bf16* A, const bf16* Bm, void* C, int c_mode,
                        int M, int N, int K,
                        long a_rs, long b_ns, long c_rs,
                        long a_bs, long b_bs, long c_bs, int batch)
{
    dim3 grid((N + 63) / 64, (M + 63) / 64, batch);
    gemm_bf16_kernel<<<grid, dim3(128), 0, s>>>(A, Bm, C, c_mode, M, N, K,
                                                a_rs, b_ns, c_rs, a_bs, b_bs, c_bs);
}

static inline dim3 ew_grid(long n) { return dim3((unsigned)((n + 255) / 256)); }

extern "C" void kernel_launch(void* const* d_in, const int* in_sizes, int n_in,
                              void* d_out, int out_size, void* d_ws, size_t ws_size,
                              hipStream_t stream)
{
    (void)in_sizes; (void)n_in; (void)out_size; (void)ws_size;

    const float* x = (const float*)d_in[0];
    // projections in (qk_q, qk_m, qk_k, qks_q, qks_m, qks_k, vo_v, vo_m) order:
    // A at d_in[1+2i], Bw at d_in[2+2i], i=0..7
    const float* w_o       = (const float*)d_in[17];
    const float* sm_lam[3] = { (const float*)d_in[18], (const float*)d_in[21], (const float*)d_in[24] };
    const float* sm_U[3]   = { (const float*)d_in[19], (const float*)d_in[22], (const float*)d_in[25] };
    const float* sm_V[3]   = { (const float*)d_in[20], (const float*)d_in[23], (const float*)d_in[26] };

    // ---- workspace carve-out ----
    char* base = (char*)d_ws;
    size_t off = 0;
    auto alloc = [&](size_t nbytes) -> void* {
        off = (off + 255) & ~(size_t)255;
        void* p = base + off;
        off += nbytes;
        return p;
    };
    bf16*  xb     = (bf16*) alloc((size_t)NX * 2);
    bf16*  Apack  = (bf16*) alloc((size_t)8 * CD * CRF * 2);
    bf16*  BpackT = (bf16*) alloc((size_t)8 * CRF * CHD * 2);  // Bw^T per proj
    bf16*  Wt     = (bf16*) alloc((size_t)8 * CD * CHD * 2);   // (A@Bw)^T per proj
    float* Y      = (float*)alloc((size_t)8 * NF * 4);
    bf16*  QM     = (bf16*) alloc((size_t)2 * NF * 2);
    bf16*  Kp     = (bf16*) alloc((size_t)2 * NF * 2);
    bf16*  Vfb    = (bf16*) alloc((size_t)NF * 2);             // [b,h,d,t]
    float* MVO    = (float*)alloc((size_t)NF * 4);
    float* S      = (float*)alloc((size_t)2 * NZ * 4);         // scores, both variants
    bf16*  Zb     = (bf16*) alloc((size_t)2 * NZ * 2);         // lag tensors, bf16
    bf16*  Pb     = (bf16*) alloc((size_t)32 * CT * CRS * 2);
    float* Q2     = (float*)alloc((size_t)NZ * 4);
    float* h1     = (float*)alloc((size_t)NZ * 4);
    float* h1s    = (float*)alloc((size_t)NZ * 4);
    bf16*  Ut     = (bf16*) alloc((size_t)CT * CRS * 2);       // U^T [RS][T]
    bf16*  Vbm    = (bf16*) alloc((size_t)CT * CRS * 2);       // V   [T][RS]
    bf16*  wob    = (bf16*) alloc((size_t)CD * CHD * 2);
    float* ctxb   = (float*)alloc((size_t)NF * 4);
    bf16*  interb = (bf16*) alloc((size_t)NF * 2);
    // aliases (lifetimes are disjoint, stream-ordered):
    float* Hf  = S;          // scores dead after seqmix combines
    bf16*  Hb  = Zb;         // lag bf16 dead after its P-GEMMs
    bf16*  RWb = (bf16*)h1;  // h1 dead after gate

    // ---- 1. convert x and low-rank weights to bf16 (Bw transposed) ----
    cvt_bf16_kernel<<<ew_grid(NX), 256, 0, stream>>>(x, xb, NX);
    for (int i = 0; i < 8; ++i) {
        cvt_bf16_kernel<<<ew_grid((long)CD * CRF), 256, 0, stream>>>(
            (const float*)d_in[1 + 2 * i], Apack + (long)i * CD * CRF, (long)CD * CRF);
        cvtT_bf16_kernel<<<ew_grid((long)CRF * CHD), 256, 0, stream>>>(
            (const float*)d_in[2 + 2 * i], BpackT + (long)i * CRF * CHD, CRF, CHD);
    }

    // ---- 2. fuse weights: Wt[i] = (A[i] @ Bw[i])^T, bf16-transposed store ----
    launch_gemm(stream, Apack, BpackT, Wt, 2, CD, CHD, CRF,
                CRF, CRF, CD,
                (long)CD * CRF, (long)CRF * CHD, (long)CD * CHD, 8);

    // ---- 3. features: Y[i] = xb @ W[i]  (8x: 2048x1024, K=1024), f32 ----
    launch_gemm(stream, xb, Wt, Y, 0, CB * CT, CHD, CD,
                CD, CD, CHD,
                0, (long)CD * CHD, NF, 8);

    // ---- 4. gated features ----
    prep_feats_kernel<<<ew_grid(NF), 256, 0, stream>>>(Y, QM, Kp, Vfb, MVO);

    // ---- 5. scores: S[b,h] = (q*mg) @ k^T  (64 batches: 1024x1024, K=64) ----
    launch_gemm(stream, QM, Kp, S, 0, CT, CT, CDH,
                CDH, CDH, CT,
                (long)CT * CDH, (long)CT * CDH, (long)CT * CT, 64);

    // ---- 6. causal-lag relabel (bf16) ----
    lag_kernel<<<ew_grid(2 * NZ), 256, 0, stream>>>(S, Zb);

    // ---- 7. seqmix helper: P = z@U (bf16), Q2 = P@V^T (f32), combine ----
    auto seqmix = [&](const bf16* zb, const float* zsrc, int gather, int smi,
                      float* outf, bf16* outb) {
        cvtT_bf16_kernel<<<ew_grid((long)CT * CRS), 256, 0, stream>>>(sm_U[smi], Ut, CT, CRS);
        cvt_bf16_kernel<<<ew_grid((long)CT * CRS), 256, 0, stream>>>(sm_V[smi], Vbm, (long)CT * CRS);
        // P = z @ U : 32 batches, 1024x32, K=1024 -> bf16
        launch_gemm(stream, zb, Ut, Pb, 1, CT, CRS, CT,
                    CT, CT, CRS,
                    (long)CT * CT, 0, (long)CT * CRS, 32);
        // Q2 = P @ V^T : 32 batches, 1024x1024, K=32 -> f32
        launch_gemm(stream, Pb, Vbm, Q2, 0, CT, CT, CRS,
                    CRS, CRS, CT,
                    (long)CT * CRS, 0, (long)CT * CT, 32);
        seqmix_combine_kernel<<<ew_grid(NZ), 256, 0, stream>>>(
            zsrc, gather, Q2, sm_lam[smi], outf, outb);
    };

    // sm1 on z (variant 0) -> h1, then row-normalize
    seqmix(Zb, S, 1, 0, h1, (bf16*)nullptr);
    rownorm_kernel<<<dim3((unsigned)(CB * CH * CT)), 256, 0, stream>>>(h1);
    // sm1s on zs (variant 1) -> h1s
    seqmix(Zb + NZ, S + NZ, 1, 1, h1s, (bf16*)nullptr);

    // ---- 8. h = softplus(h1s) * relu(h1n) ----
    gate_kernel<<<ew_grid(NZ), 256, 0, stream>>>(h1, h1s, Hf, Hb);

    // ---- 9. rw = seqmix(h) -> bf16 only ----
    seqmix(Hb, Hf, 0, 2, (float*)nullptr, RWb);

    // ---- 10. ctx = rw @ v : 32 batches, 1024x64, K=1024 (B = v^T layout) ----
    launch_gemm(stream, RWb, Vfb, ctxb, 0, CT, CDH, CT,
                CT, CT, CDH,
                (long)CT * CT, (long)CDH * CT, (long)CT * CDH, 32);

    // ---- 11. inter = ctx * mvo, packed [B*T, H*DH] ----
    inter_kernel<<<ew_grid(NF), 256, 0, stream>>>(ctxb, MVO, interb);

    // ---- 12. out = inter @ w_o^T : 2048x1024, K=1024 -> f32 d_out ----
    cvt_bf16_kernel<<<ew_grid((long)CD * CHD), 256, 0, stream>>>(w_o, wob, (long)CD * CHD);
    launch_gemm(stream, interb, wob, d_out, 0, CB * CT, CD, CHD,
                CHD, CHD, CD,
                0, 0, 0, 1);
}
